// MaskLayer_32229434589395
// MI455X (gfx1250) — compile-verified, matrix operand order checked by reading, not run on hardware
//
#include <hip/hip_runtime.h>
#include <hip/hip_bf16.h>

// ---- types for WMMA ----
typedef _Float16 v16h __attribute__((ext_vector_type(16)));
typedef _Float16 v4h  __attribute__((ext_vector_type(4)));
typedef float    v8f  __attribute__((ext_vector_type(8)));
typedef float    v4f  __attribute__((ext_vector_type(4)));
typedef unsigned int v4u __attribute__((ext_vector_type(4)));

#define SZ    32            // crop size
#define PD    34            // padded tile (zero halo for SAME conv)
#define CHN   64            // channels
#define NBOX  1024
#define FRAG_PER_CONV (9*2*4*32*16)   // [tap][kstep][ntile][lane][16 halves]

// ---------------------------------------------------------------------------
// Pack conv1_w / conv2_w (3,3,64,64 HWIO, fp32) into f16 WMMA B-fragments.
// ---------------------------------------------------------------------------
__global__ void mask_prep_weights(const float* __restrict__ w1,
                                  const float* __restrict__ w2,
                                  _Float16* __restrict__ frag) {
    int i = blockIdx.x * blockDim.x + threadIdx.x;
    const int total = 2 * FRAG_PER_CONV;
    if (i >= total) return;
    int h    = i & 15;
    int r    = i >> 4;
    int lane = r & 31;  r >>= 5;
    int nt   = r & 3;   r >>= 2;
    int ks   = r & 1;   r >>= 1;
    int tap  = r % 9;
    int conv = r / 9;
    int lg   = lane >> 4;
    int K = ks * 32 + ((h < 8) ? (h + 8 * lg) : (16 + (h - 8) + 8 * lg));
    int N = nt * 16 + (lane & 15);
    const float* w = conv ? w2 : w1;
    frag[i] = (_Float16)w[(tap * 64 + K) * 64 + N];   // HWIO flat
}

// ---------------------------------------------------------------------------
// 3x3 SAME conv 64->64 as implicit GEMM with v_wmma_f32_16x16x32_f16.
// Wave owns 8 M-tiles, processed as 2 passes of 4:
//   acc[4][4]    = 128 VGPRs live across the tap loop
//   B frags b[4] =  32 VGPRs, loaded once per (tap,ks), reused by 4 M-tiles
// Tap loop is NOT unrolled; only the 32-bit byte OFFSET of the weight
// fragments goes through an asm opacity barrier. This (a) prevents LICM
// from hoisting all 72 fragments (scratch spills in rev 1), and (b) keeps
// the base pointer typed/global so loads lower to global_load_b128 (GVS)
// instead of flat_load_b128 (rev 2), decoupling LOADcnt from DScnt.
// ---------------------------------------------------------------------------
__device__ __forceinline__ void conv3x3_wmma(const _Float16* __restrict__ src,
                                             _Float16* __restrict__ dst,
                                             const _Float16* __restrict__ wfrag,
                                             const float* __restrict__ bias,
                                             int tid) {
    const int lane = tid & 31;
    const int wave = tid >> 5;
    const int lg   = lane >> 4;     // lane group (K-half select)
    const int ml   = lane & 15;     // M (A) / N (B,C) within tile
    float bs[4];
#pragma unroll
    for (int nt = 0; nt < 4; ++nt) bs[nt] = bias[nt * 16 + ml];

    const char* wbase = (const char*)wfrag;

#pragma unroll 1
    for (int pass = 0; pass < 2; ++pass) {
        const int mt0 = wave * 8 + pass * 4;
        v8f acc[4][4] = {};   // [m][ntile]

#pragma unroll 1
        for (int tap = 0; tap < 9; ++tap) {
            const int dy = tap / 3, dx = tap % 3;
#pragma unroll
            for (int ks = 0; ks < 2; ++ks) {
                // --- load 4 B fragments for this (tap,ks), keep in VGPRs ---
                unsigned woff = (unsigned)(((tap * 2 + ks) * 4 * 32 * 16
                                            + lane * 16) * sizeof(_Float16));
                asm volatile("" : "+v"(woff));   // defeat LICM, keep base global
                union { v16h h; v4u u[2]; } b[4];
#pragma unroll
                for (int nt = 0; nt < 4; ++nt) {
                    const v4u* q = (const v4u*)(wbase + woff + nt * 1024);
                    b[nt].u[0] = q[0];           // global_load_b128
                    b[nt].u[1] = q[1];           // global_load_b128
                }
                // --- 4 M-tiles share these B fragments ---
#pragma unroll
                for (int m = 0; m < 4; ++m) {
                    const int p = (mt0 + m) * 16 + ml;
                    const int y = p >> 5, x = p & 31;
                    const _Float16* abase =
                        src + ((y + dy) * PD + (x + dx)) * CHN;
                    union { v16h h; v4u u[2]; } a;
                    const int kb = ks * 32 + 8 * lg;
                    a.u[0] = *(const v4u*)(abase + kb);        // ds_load_b128
                    a.u[1] = *(const v4u*)(abase + kb + 16);   // ds_load_b128
#pragma unroll
                    for (int nt = 0; nt < 4; ++nt)
                        acc[m][nt] = __builtin_amdgcn_wmma_f32_16x16x32_f16(
                            false, a.h, false, b[nt].h, (short)0,
                            acc[m][nt], false, false);
                }
            }
        }

        // bias + ReLU, store f16 into dst interior
#pragma unroll
        for (int m = 0; m < 4; ++m) {
#pragma unroll
            for (int nt = 0; nt < 4; ++nt) {
#pragma unroll
                for (int e = 0; e < 8; ++e) {
                    float v = fmaxf(acc[m][nt][e] + bs[nt], 0.0f);
                    const int M  = (mt0 + m) * 16 + e + 8 * lg;
                    const int yy = M >> 5, xx = M & 31;
                    dst[((yy + 1) * PD + (xx + 1)) * CHN + nt * 16 + ml] =
                        (_Float16)v;
                }
            }
        }
    }
}

// ---------------------------------------------------------------------------
// One workgroup per box: crop -> conv1 -> conv2 (WMMA) -> class-selected
// VALID conv; fully LDS-resident (2 x 34x34x64 f16 = 289 KB of 320 KB WGP LDS).
// ---------------------------------------------------------------------------
__global__ __launch_bounds__(256) void mask_fused_kernel(
    const float* __restrict__ features,   // [8,192,192,64]
    const float* __restrict__ boxes,      // [N,4]
    const int*   __restrict__ box_ids,    // [N]
    const int*   __restrict__ cls,        // [N]
    const float* __restrict__ b1,
    const float* __restrict__ b2,
    const float* __restrict__ out_w,      // [3,3,64,3]
    const float* __restrict__ out_b,      // [3]
    const _Float16* __restrict__ wfrag,   // packed weights (2 convs)
    float* __restrict__ out)              // [N,30,30,1]
{
    extern __shared__ char smem_raw[];
    _Float16* padA = (_Float16*)smem_raw;       // PD*PD*CHN halves
    _Float16* padB = padA + PD * PD * CHN;

    const int tid = threadIdx.x;
    const int n   = blockIdx.x;

    // ---- stage 0: zero LDS (free zero halo for SAME convs) ----
    {
        v4u z = {0, 0, 0, 0};
        v4u* p = (v4u*)smem_raw;
        const int nvec = 2 * PD * PD * CHN / 8;
        for (int i = tid; i < nvec; i += 256) p[i] = z;
    }
    __syncthreads();

    // ---- stage 1: bilinear crop_and_resize into padA interior ----
    {
        const float y1 = boxes[4 * n + 0], x1 = boxes[4 * n + 1];
        const float y2 = boxes[4 * n + 2], x2 = boxes[4 * n + 3];
        const int bi = box_ids[n];
        const float* fb = features + (size_t)bi * 192 * 192 * 64;
        for (int p = tid; p < SZ * SZ; p += 256) {
            const int oy = p >> 5, ox = p & 31;
            const float ys = (y1 + (oy * (1.0f / 31.0f)) * (y2 - y1)) * 191.0f;
            const float xs = (x1 + (ox * (1.0f / 31.0f)) * (x2 - x1)) * 191.0f;
            float y0f = fminf(fmaxf(floorf(ys), 0.0f), 191.0f);
            float x0f = fminf(fmaxf(floorf(xs), 0.0f), 191.0f);
            const float wy = ys - y0f, wx = xs - x0f;
            const int y0i = (int)y0f, x0i = (int)x0f;
            const int y1i = (y0i + 1 > 191) ? 191 : y0i + 1;
            const int x1i = (x0i + 1 > 191) ? 191 : x0i + 1;
            const v4f* f00 = (const v4f*)(fb + (y0i * 192 + x0i) * 64);
            const v4f* f01 = (const v4f*)(fb + (y0i * 192 + x1i) * 64);
            const v4f* f10 = (const v4f*)(fb + (y1i * 192 + x0i) * 64);
            const v4f* f11 = (const v4f*)(fb + (y1i * 192 + x1i) * 64);
            _Float16* dst = padA + ((oy + 1) * PD + (ox + 1)) * CHN;
#pragma unroll 4
            for (int c = 0; c < CHN / 4; ++c) {
                v4f v00 = f00[c], v01 = f01[c], v10 = f10[c], v11 = f11[c];
                v4f top = v00 + (v01 - v00) * wx;
                v4f bot = v10 + (v11 - v10) * wx;
                v4f r   = top + (bot - top) * wy;
                v4h h;
                h.x = (_Float16)r.x; h.y = (_Float16)r.y;
                h.z = (_Float16)r.z; h.w = (_Float16)r.w;
                *(v4h*)(dst + 4 * c) = h;   // 8B LDS store
            }
        }
    }
    __syncthreads();

    // ---- stage 2: conv1 (padA -> padB) ----
    conv3x3_wmma(padA, padB, wfrag, b1, tid);
    __syncthreads();

    // ---- stage 3: conv2 (padB -> padA) ----
    conv3x3_wmma(padB, padA, wfrag + FRAG_PER_CONV, b2, tid);
    __syncthreads();

    // ---- stage 4a: stage selected-class out_w into LDS (padB is free) ----
    float* wsel = (float*)padB;
    const int c = cls[n];
    for (int i = tid; i < 9 * 64; i += 256) wsel[i] = out_w[i * 3 + c];
    __syncthreads();

    // ---- stage 4b: 3x3 VALID conv, one output channel, 30x30 ----
    {
        const float ob = out_b[c];
        for (int p = tid; p < 30 * 30; p += 256) {
            const int oy = p / 30, ox = p % 30;
            float acc = ob;
#pragma unroll
            for (int tap = 0; tap < 9; ++tap) {
                const int dy = tap / 3, dx = tap % 3;
                const _Float16* xb =
                    padA + ((oy + dy + 1) * PD + (ox + dx + 1)) * CHN;
                const float* wv = wsel + tap * 64;
#pragma unroll
                for (int k8 = 0; k8 < 8; ++k8) {
                    union { v4u u; _Float16 f[8]; } t;
                    t.u = *(const v4u*)(xb + k8 * 8);   // ds_load_b128
#pragma unroll
                    for (int j = 0; j < 8; ++j)
                        acc += (float)t.f[j] * wv[k8 * 8 + j];
                }
            }
            out[n * 900 + p] = acc;
        }
    }
}

extern "C" void kernel_launch(void* const* d_in, const int* in_sizes, int n_in,
                              void* d_out, int out_size, void* d_ws, size_t ws_size,
                              hipStream_t stream) {
    const float* features = (const float*)d_in[0];
    const float* boxes    = (const float*)d_in[1];
    const int*   box_ids  = (const int*)d_in[2];
    const int*   cls      = (const int*)d_in[3];
    const float* w1       = (const float*)d_in[4];
    const float* b1       = (const float*)d_in[5];
    const float* w2       = (const float*)d_in[6];
    const float* b2       = (const float*)d_in[7];
    const float* ow       = (const float*)d_in[8];
    const float* ob       = (const float*)d_in[9];
    _Float16* wfrag = (_Float16*)d_ws;   // 2*FRAG_PER_CONV halves = 144 KB

    const int total = 2 * FRAG_PER_CONV;
    mask_prep_weights<<<(total + 255) / 256, 256, 0, stream>>>(w1, w2, wfrag);

    const size_t lds_bytes = (size_t)2 * PD * PD * CHN * sizeof(_Float16);
    mask_fused_kernel<<<NBOX, 256, lds_bytes, stream>>>(
        features, boxes, box_ids, cls, b1, b2, ow, ob, wfrag, (float*)d_out);
}
